// AttentionWithBias_87222195848110
// MI455X (gfx1250) — compile-verified, matrix operand order checked by reading, not run on hardware
//
#include <hip/hip_runtime.h>
#include <hip/hip_bf16.h>

typedef __attribute__((ext_vector_type(16))) _Float16 v16h;
typedef __attribute__((ext_vector_type(8)))  _Float16 v8h;
typedef __attribute__((ext_vector_type(8)))  float    v8f;
typedef __attribute__((ext_vector_type(4)))  int      v4i;

#define D_MODEL 1024
#define HEADS   16
#define HD      64
#define BATCH   4
#define SEQ     1024

#if __has_builtin(__builtin_amdgcn_global_load_async_to_lds_b128) && \
    __has_builtin(__builtin_amdgcn_s_wait_asynccnt)
#define HAVE_ASYNC_LDS 1
#else
#define HAVE_ASYNC_LDS 0
#endif

// global / LDS address-space v4i pointers for the async-copy builtin
#define AS1P(p) ((__attribute__((address_space(1))) v4i*)(p))
#define AS3P(p) ((__attribute__((address_space(3))) v4i*)(p))

__device__ __forceinline__ v8f wmma16x16x32(v16h a, v16h b, v8f c) {
  // v_wmma_f32_16x16x32_f16: D = A*B + C, f32 accum
  return __builtin_amdgcn_wmma_f32_16x16x32_f16(false, a, false, b, (short)0, c, false, false);
}

// Load a 16x32 f16 A/B fragment from a row-major tile.
// Lane l: row = l&15, K = (l>>4)*8 + {0..7} and 16 + (l>>4)*8 + {0..7}
__device__ __forceinline__ v16h load_frag(const _Float16* p, int rowstride, int lane) {
  int r  = lane & 15;
  int ko = (lane >> 4) << 3;
  const _Float16* q = p + (size_t)r * rowstride + ko;
  v8h lo = *(const v8h*)(q);
  v8h hi = *(const v8h*)(q + 16);
  v16h out;
#pragma unroll
  for (int i = 0; i < 8; ++i) { out[i] = lo[i]; out[i + 8] = hi[i]; }
  return out;
}

__device__ __forceinline__ float rowmax16(float v) {
#pragma unroll
  for (int m = 1; m < 16; m <<= 1) v = fmaxf(v, __shfl_xor(v, m, 32));
  return v;
}
__device__ __forceinline__ float rowsum16(float v) {
#pragma unroll
  for (int m = 1; m < 16; m <<= 1) v += __shfl_xor(v, m, 32);
  return v;
}

__global__ void cvt_f32_f16(const float* __restrict__ in, _Float16* __restrict__ out, int n) {
  int i = blockIdx.x * blockDim.x + threadIdx.x;
  if (i < n) out[i] = (_Float16)in[i];
}

// Stage one 64(N) x 32(K) f16 W tile (4 KB) into LDS, block-cooperatively.
// 256 16-byte chunks, 128 threads x 2. Async global->LDS when available.
__device__ __forceinline__ void stage_B(const _Float16* __restrict__ W,
                                        _Float16* buf, int n0, int kb, int tid) {
#pragma unroll
  for (int i = 0; i < 2; ++i) {
    int c   = tid + i * 128;
    int row = c >> 2;
    int q   = (c & 3) * 8;
    const _Float16* gp = W + (size_t)(n0 + row) * D_MODEL + kb + q;
    _Float16* lp = buf + row * 32 + q;
#if HAVE_ASYNC_LDS
    __builtin_amdgcn_global_load_async_to_lds_b128(AS1P(gp), AS3P(lp), 0, 0);
#else
    *(v8h*)lp = *(const v8h*)gp;
#endif
  }
}

// Double-buffered GEMM core: one wave's 16(M) x 64(N) tile, K = 1024.
// B tile shared by the 4 waves of the block via LDS; A register-double-buffered.
__device__ __forceinline__ void gemm_core(const _Float16* __restrict__ A,
                                          const _Float16* __restrict__ W,
                                          _Float16* bshare,   // [2][64*32]
                                          int m0, int n0, int lane, int tid,
                                          v8f acc[4]) {
  stage_B(W, bshare, n0, 0, tid);
  v16h a = load_frag(A + (size_t)m0 * D_MODEL, D_MODEL, lane);
  for (int kb = 0; kb < D_MODEL; kb += 32) {
    _Float16* cur = bshare + ((kb >> 5) & 1) * (64 * 32);
    _Float16* nxt = bshare + (((kb >> 5) + 1) & 1) * (64 * 32);
    if (kb + 32 < D_MODEL) {
      stage_B(W, nxt, n0, kb + 32, tid);   // fill next buffer in flight
#if HAVE_ASYNC_LDS
      __builtin_amdgcn_s_wait_asynccnt(2); // cur buffer's 2 chunks landed (in-order)
#endif
    } else {
#if HAVE_ASYNC_LDS
      __builtin_amdgcn_s_wait_asynccnt(0);
#endif
    }
    __syncthreads();
    v16h a_cur = a;
    if (kb + 32 < D_MODEL)
      a = load_frag(A + (size_t)m0 * D_MODEL + kb + 32, D_MODEL, lane);
#pragma unroll
    for (int t = 0; t < 4; ++t) {
      v16h bfrag = load_frag(cur + t * 16 * 32, 32, lane);
      acc[t] = wmma16x16x32(a_cur, bfrag, acc[t]);
    }
    __syncthreads();
  }
}

// Fused QKV projection: C = xh @ W^T; z selects Wq/Wk/Wv and output layout.
__global__ void qkv_gemm(const _Float16* __restrict__ xh,
                         const _Float16* __restrict__ Wq,
                         const _Float16* __restrict__ Wk,
                         const _Float16* __restrict__ Wv,
                         _Float16* __restrict__ Qb,
                         _Float16* __restrict__ Kb,
                         _Float16* __restrict__ Vt) {
  __shared__ alignas(16) _Float16 bshare[2 * 64 * 32];
  int tid  = threadIdx.x;
  int lane = tid & 31;
  int wave = tid >> 5;
  int m0 = (blockIdx.x * 4 + wave) * 16;
  int n0 = blockIdx.y * 64;
  const _Float16* W; _Float16* dst; int vmode;
  if (blockIdx.z == 0)      { W = Wq; dst = Qb; vmode = 0; }
  else if (blockIdx.z == 1) { W = Wk; dst = Kb; vmode = 0; }
  else                      { W = Wv; dst = Vt; vmode = 1; }

  v8f acc[4] = {{}, {}, {}, {}};
  gemm_core(xh, W, bshare, m0, n0, lane, tid, acc);

  int hi8 = (lane & 16) ? 8 : 0;
#pragma unroll
  for (int t = 0; t < 4; ++t) {
#pragma unroll
    for (int r = 0; r < 8; ++r) {
      int gm = m0 + r + hi8;
      int gc = n0 + t * 16 + (lane & 15);
      int bb = gm >> 10, npos = gm & 1023;
      int h  = gc >> 6,  hc   = gc & 63;
      size_t idx = vmode
        ? (((size_t)(bb * HEADS + h) * HD + hc) * SEQ + npos)   // Vt[b,h,hd,N]
        : (((size_t)(bb * HEADS + h) * SEQ + npos) * HD + hc);  // Q/K[b,h,N,hd]
      dst[idx] = (_Float16)acc[t][r];
    }
  }
}

// out = O @ Wo^T, f32 output
__global__ void oproj_gemm(const _Float16* __restrict__ Ob,
                           const _Float16* __restrict__ Wo,
                           float* __restrict__ out) {
  __shared__ alignas(16) _Float16 bshare[2 * 64 * 32];
  int tid  = threadIdx.x;
  int lane = tid & 31;
  int wave = tid >> 5;
  int m0 = (blockIdx.x * 4 + wave) * 16;
  int n0 = blockIdx.y * 64;
  v8f acc[4] = {{}, {}, {}, {}};
  gemm_core(Ob, Wo, bshare, m0, n0, lane, tid, acc);
  int hi8 = (lane & 16) ? 8 : 0;
#pragma unroll
  for (int t = 0; t < 4; ++t)
#pragma unroll
    for (int r = 0; r < 8; ++r)
      out[(size_t)(m0 + r + hi8) * D_MODEL + n0 + t * 16 + (lane & 15)] = acc[t][r];
}

// Flash attention: one wave = 16 queries x all 1024 keys for one (b,h).
__global__ void attn_kernel(const _Float16* __restrict__ Q,
                            const _Float16* __restrict__ K,
                            const _Float16* __restrict__ Vt,
                            const float* __restrict__ bias,
                            _Float16* __restrict__ O) {
  __shared__ alignas(16) _Float16 plds[4][16 * 32];
  int lane = threadIdx.x & 31;
  int wib  = threadIdx.x >> 5;
  int w  = blockIdx.x * 4 + wib;
  int bh = w >> 6;
  int q0 = (w & 63) << 4;
  int b = bh >> 4, h = bh & 15;
  const _Float16* Qh = Q  + (size_t)bh * SEQ * HD;
  const _Float16* Kh = K  + (size_t)bh * SEQ * HD;
  const _Float16* Vh = Vt + (size_t)bh * HD * SEQ;
  const float* bbias = bias + (size_t)bh * SEQ * SEQ;
  int col = lane & 15;
  int hi8 = (lane & 16) ? 8 : 0;
  const float scale = 0.125f;  // hd^-0.5

  v16h a0 = load_frag(Qh + (size_t)q0 * HD + 0,  HD, lane);
  v16h a1 = load_frag(Qh + (size_t)q0 * HD + 32, HD, lane);

  v8f acc[4] = {{}, {}, {}, {}};
  float mrow[8], lrow[8];
#pragma unroll
  for (int r = 0; r < 8; ++r) { mrow[r] = -3.0e38f; lrow[r] = 0.f; }

  for (int jb = 0; jb < SEQ; jb += 32) {
    // S tiles: 16 queries x 32 keys, K-dim = hd = 64
    v8f s0 = {}, s1 = {};
    {
      v16h bk = load_frag(Kh + (size_t)jb * HD + 0,  HD, lane);
      s0 = wmma16x16x32(a0, bk, s0);
      bk = load_frag(Kh + (size_t)jb * HD + 32, HD, lane);
      s0 = wmma16x16x32(a1, bk, s0);
      bk = load_frag(Kh + (size_t)(jb + 16) * HD + 0,  HD, lane);
      s1 = wmma16x16x32(a0, bk, s1);
      bk = load_frag(Kh + (size_t)(jb + 16) * HD + 32, HD, lane);
      s1 = wmma16x16x32(a1, bk, s1);
    }
    // V fragments issued early so their latency hides under the softmax VALU work
    v16h bv[4];
#pragma unroll
    for (int t = 0; t < 4; ++t)
      bv[t] = load_frag(Vh + (size_t)(t * 16) * SEQ + jb, SEQ, lane);

    // scale + bias (read-once: non-temporal)
#pragma unroll
    for (int r = 0; r < 8; ++r) {
      size_t row = (size_t)(q0 + r + hi8) * SEQ;
      s0[r] = s0[r] * scale + __builtin_nontemporal_load(bbias + row + jb + col);
      s1[r] = s1[r] * scale + __builtin_nontemporal_load(bbias + row + jb + 16 + col);
    }
    // online softmax update per row
    float p0[8], p1[8];
#pragma unroll
    for (int r = 0; r < 8; ++r) {
      float mn = fmaxf(mrow[r], rowmax16(fmaxf(s0[r], s1[r])));
      float f  = __expf(mrow[r] - mn);
      mrow[r] = mn;
      p0[r] = __expf(s0[r] - mn);
      p1[r] = __expf(s1[r] - mn);
      lrow[r] = lrow[r] * f + rowsum16(p0[r] + p1[r]);
#pragma unroll
      for (int t = 0; t < 4; ++t) acc[t][r] *= f;
    }
    // C-layout -> A-layout for P via per-wave LDS tile
    _Float16* my = plds[wib];
#pragma unroll
    for (int r = 0; r < 8; ++r) {
      int rr = r + hi8;
      my[rr * 32 + col]      = (_Float16)p0[r];
      my[rr * 32 + 16 + col] = (_Float16)p1[r];
    }
    __syncthreads();
    v16h pa = load_frag(my, 32, lane);
#pragma unroll
    for (int t = 0; t < 4; ++t)
      acc[t] = wmma16x16x32(pa, bv[t], acc[t]);
    __syncthreads();
  }
#pragma unroll
  for (int r = 0; r < 8; ++r) {
    float inv = 1.0f / lrow[r];
    size_t row = (size_t)(b * SEQ + q0 + r + hi8) * D_MODEL;
#pragma unroll
    for (int t = 0; t < 4; ++t)
      O[row + h * HD + t * 16 + col] = (_Float16)(acc[t][r] * inv);
  }
}

extern "C" void kernel_launch(void* const* d_in, const int* in_sizes, int n_in,
                              void* d_out, int out_size, void* d_ws, size_t ws_size,
                              hipStream_t stream) {
  (void)in_sizes; (void)n_in; (void)out_size; (void)ws_size;
  const float* x    = (const float*)d_in[0];
  const float* bias = (const float*)d_in[1];
  const float* Wq   = (const float*)d_in[2];
  const float* Wk   = (const float*)d_in[3];
  const float* Wv   = (const float*)d_in[4];
  const float* Wo   = (const float*)d_in[5];

  const size_t NX = (size_t)BATCH * SEQ * D_MODEL;  // 4,194,304
  const size_t NW = (size_t)D_MODEL * D_MODEL;      // 1,048,576
  _Float16* ws  = (_Float16*)d_ws;
  _Float16* xh  = ws;             // f16 x (reused as attention output Ob)
  _Float16* Wqh = xh  + NX;
  _Float16* Wkh = Wqh + NW;
  _Float16* Wvh = Wkh + NW;
  _Float16* Woh = Wvh + NW;
  _Float16* Qb  = Woh + NW;       // [B,H,N,hd]
  _Float16* Kb  = Qb  + NX;       // [B,H,N,hd]
  _Float16* Vtb = Kb  + NX;       // [B,H,hd,N]
  _Float16* Ob  = xh;             // xh dead after QKV GEMM

  cvt_f32_f16<<<(int)(NX / 256), 256, 0, stream>>>(x,  xh,  (int)NX);
  cvt_f32_f16<<<(int)(NW / 256), 256, 0, stream>>>(Wq, Wqh, (int)NW);
  cvt_f32_f16<<<(int)(NW / 256), 256, 0, stream>>>(Wk, Wkh, (int)NW);
  cvt_f32_f16<<<(int)(NW / 256), 256, 0, stream>>>(Wv, Wvh, (int)NW);
  cvt_f32_f16<<<(int)(NW / 256), 256, 0, stream>>>(Wo, Woh, (int)NW);

  dim3 gq(64, 16, 3);
  qkv_gemm<<<gq, 128, 0, stream>>>(xh, Wqh, Wkh, Wvh, Qb, Kb, Vtb);

  attn_kernel<<<1024, 128, 0, stream>>>(Qb, Kb, Vtb, bias, Ob);

  dim3 go(64, 16, 1);
  oproj_gemm<<<go, 128, 0, stream>>>(Ob, Woh, (float*)d_out);
}
